// BERT_BiGRU_STACK_CNN_CRF_11269994185175
// MI455X (gfx1250) — matrix operand order repaired and checked by reading, not compiled, not run on hardware
//
#include <hip/hip_runtime.h>
#include <hip/hip_bf16.h>

// ---------------- problem constants ----------------
#define B_    64
#define S_    512
#define D_    768
#define H_    64
#define G3_   192          // 3*H
#define SP_   (S_ + 62)    // padded sequence length (31 each side)
#define KTOT_ (63 * D_)    // dense conv K = 48384
#define NL_   9            // CRF labels
#define KB_   128          // conv K values staged per LDS stage
#define NSTAGE_ (KTOT_ / KB_)   // 378

typedef __attribute__((ext_vector_type(16))) _Float16     v16h;
typedef __attribute__((ext_vector_type(8)))  float        v8f;
typedef __attribute__((ext_vector_type(4)))  float        f32x4;
typedef __attribute__((ext_vector_type(4)))  unsigned int u32x4;
typedef __attribute__((ext_vector_type(8)))  int          i32x8;
typedef __attribute__((ext_vector_type(4)))  int          i32x4;

struct Ptr32 { const float* p[32]; };

// ---------------- WMMA helpers (CDNA5 layouts) ----------------
// A (16x32 f16): lane L<16 holds row L, K in {0..7, 16..23}; lane L+16 holds
// row L, K in {8..15, 24..31}. Two b128 loads from row-major memory.
__device__ __forceinline__ v16h load_a16(const _Float16* tile, int lda, int lane) {
    const int row  = lane & 15;
    const int koff = (lane & 16) ? 8 : 0;
    const _Float16* p = tile + (size_t)row * lda + koff;
    v16h a;
    f32x4* av = reinterpret_cast<f32x4*>(&a);
    av[0] = *reinterpret_cast<const f32x4*>(p);        // K = koff .. koff+7
    av[1] = *reinterpret_cast<const f32x4*>(p + 16);   // K = koff+16 .. koff+23
    return a;
}

// B (32x16 f16): lane L<16 holds col L, K in {0..7, 16..23}; lane L+16 holds
// col L-16, K in {8..15, 24..31} (matches A). W stored row-major as (N, K).
__device__ __forceinline__ v16h load_b16(const _Float16* tile, int ldb, int lane) {
    const int col   = lane & 15;
    const int kbase = (lane & 16) ? 8 : 0;
    const _Float16* p = tile + (size_t)col * ldb + kbase;
    v16h b;
    f32x4* bv = reinterpret_cast<f32x4*>(&b);
    bv[0] = *reinterpret_cast<const f32x4*>(p);
    bv[1] = *reinterpret_cast<const f32x4*>(p + 16);
    return b;
}

__device__ __forceinline__ v8f wmma16(v16h a, v16h b, v8f c) {
    return __builtin_amdgcn_wmma_f32_16x16x32_f16(false, a, false, b, (short)0, c,
                                                  false, false);
}

// ---------------- TDM staging of a 2D B-panel into LDS ----------------
// D# per CDNA5 ISA ch.8: Group0 = {count/flags, lds_addr, global_addr, type},
// Group1 = {mask/data_size, tensor dims, tile dims, dim0 stride}. 2D tensor:
// groups 2/3 unused. data_size=1 (2 bytes). Tile: KB_ x 64ch, row stride KTOT_.
#if __has_builtin(__builtin_amdgcn_tensor_load_to_lds) && \
    __has_builtin(__builtin_amdgcn_s_wait_tensorcnt)
#define USE_TDM_ 1
#else
#define USE_TDM_ 0
#endif

#if USE_TDM_
__device__ __forceinline__ void tdm_stage_b(const _Float16* gsrc, unsigned lds_off) {
    const unsigned long long ga = (unsigned long long)(uintptr_t)gsrc;
    u32x4 g0;
    g0[0] = 1u;                                        // count=1 user descriptor
    g0[1] = lds_off;                                   // lds_addr (bytes)
    g0[2] = (unsigned)(ga & 0xFFFFFFFFu);              // global_addr[31:0]
    g0[3] = (unsigned)((ga >> 32) & 0x1FFFFFFu) | (2u << 30);  // [56:32] | type=2
    i32x8 g1;
    g1[0] = 1 << 16;                                   // data_size=1 (2B), mask=0
    g1[1] = (KB_ & 0xFFFF) << 16;                      // tensor_dim0 lo16
    g1[2] = ((KB_ >> 16) & 0xFFFF) | (64 << 16);       // dim0 hi | tensor_dim1 lo
    g1[3] = 0 | (KB_ << 16);                           // dim1 hi | tile_dim0=KB_
    g1[4] = 64;                                        // tile_dim1=64, tile_dim2=0
    g1[5] = KTOT_;                                     // tensor_dim0_stride lo32
    g1[6] = 0;                                         // stride hi | dim1_stride lo
    g1[7] = 0;
    const i32x4 z4 = {0, 0, 0, 0};
    const i32x8 z8 = {0, 0, 0, 0, 0, 0, 0, 0};
    // 6-arg form (clang-23 / therock headers): (g0, g1, g2, g3, g4, cpol)
    __builtin_amdgcn_tensor_load_to_lds(g0, g1, z4, z4, z8, 0);
}
#endif

// fallback staging: registers -> LDS, same double-buffer pipeline shape
__device__ __forceinline__ void stage_fb_load(const _Float16* Wd_k0, int tid,
                                              f32x4 stg[4]) {
#pragma unroll
    for (int j = 0; j < 4; ++j) {
        int idx = (j * 256 + tid) * 8;     // 8 halves per b128
        int c = idx >> 7;                  // / KB_
        int o = idx & (KB_ - 1);
        stg[j] = *reinterpret_cast<const f32x4*>(Wd_k0 + (size_t)c * KTOT_ + o);
    }
}
__device__ __forceinline__ void stage_fb_store(_Float16* dst, int tid,
                                               const f32x4 stg[4]) {
#pragma unroll
    for (int j = 0; j < 4; ++j) {
        int idx = (j * 256 + tid) * 8;
        *reinterpret_cast<f32x4*>(dst + idx) = stg[j];
    }
}

// ---------------- pack / convert kernels ----------------
__global__ void pad_convert_x_kernel(const float* __restrict__ x,
                                     _Float16* __restrict__ xpad) {
    size_t i = (size_t)blockIdx.x * blockDim.x + threadIdx.x;
    const size_t tot = (size_t)B_ * SP_ * D_;
    if (i >= tot) return;
    int d  = (int)(i % D_);
    size_t r = i / D_;
    int sp = (int)(r % SP_);
    int b  = (int)(r / SP_);
    int s  = sp - 31;
    _Float16 v = (_Float16)0.f;
    if (s >= 0 && s < S_) v = (_Float16)x[((size_t)b * S_ + s) * D_ + d];
    xpad[i] = v;
}

// Dense conv weights: Wd[c][t*768 + d], t = offset+31, zero outside support.
__global__ void pack_wd_kernel(Ptr32 cws, _Float16* __restrict__ Wd) {
    size_t i = (size_t)blockIdx.x * blockDim.x + threadIdx.x;
    const size_t tot = (size_t)64 * KTOT_;
    if (i >= tot) return;
    int c   = (int)(i / KTOT_);
    int rem = (int)(i % KTOT_);
    int t   = rem / D_;
    int d   = rem % D_;
    int ki  = c >> 1;            // kernel size k = 2*ki+1, half-width = ki
    int o   = t - 31;
    _Float16 v = (_Float16)0.f;
    if (o >= -ki && o <= ki) {
        const float* w = cws.p[ki];  // shape (2, 768, 2*ki+1)
        v = (_Float16)w[((size_t)(c & 1) * D_ + d) * (size_t)(2 * ki + 1) + (o + ki)];
    }
    Wd[i] = v;
}

__global__ void pack_misc_kernel(const float* __restrict__ WihF,
                                 const float* __restrict__ WihB,
                                 const float* __restrict__ linW, Ptr32 cbs,
                                 _Float16* __restrict__ WihFh,
                                 _Float16* __restrict__ WihBh,
                                 _Float16* __restrict__ linWh,
                                 float* __restrict__ cbias) {
    const int NW = G3_ * D_;       // 147456
    size_t i = (size_t)blockIdx.x * blockDim.x + threadIdx.x;
    if (i < (size_t)NW) {
        WihFh[i] = (_Float16)WihF[i];
    } else if (i < (size_t)2 * NW) {
        size_t j = i - NW;
        WihBh[j] = (_Float16)WihB[j];
    } else if (i < (size_t)2 * NW + 16 * G3_) {
        size_t j = i - (size_t)2 * NW;
        int row = (int)(j / G3_), col = (int)(j % G3_);
        linWh[j] = (row < NL_) ? (_Float16)linW[(size_t)row * G3_ + col] : (_Float16)0.f;
    } else if (i < (size_t)2 * NW + 16 * G3_ + 64) {
        int c = (int)(i - ((size_t)2 * NW + 16 * G3_));
        cbias[c] = cbs.p[c >> 1][c & 1];
    }
}

// ---------------- GRU input projection: xp = x @ Wih^T + bih ----------------
__global__ __launch_bounds__(32) void gru_xproj_kernel(
    const _Float16* __restrict__ xpad, const _Float16* __restrict__ WihFh,
    const _Float16* __restrict__ WihBh, const float* __restrict__ bihF,
    const float* __restrict__ bihB, float* __restrict__ xpF,
    float* __restrict__ xpB) {
    const int mt = blockIdx.x;            // 0..2047 (M = B*S, tile of 16)
    const int nt = blockIdx.y;            // 0..11   (N = 192)
    const int dir = blockIdx.z;
    const int lane = threadIdx.x;
    const int m0 = mt * 16;
    const int b  = m0 >> 9;
    const int s0 = m0 & 511;
    const _Float16* A  = xpad + ((size_t)(b * SP_ + s0 + 31)) * D_;
    const _Float16* Bt = (dir ? WihBh : WihFh) + (size_t)(nt * 16) * D_;
    v8f c = {};
#pragma unroll 4
    for (int kc = 0; kc < D_ / 32; ++kc) {
        v16h a = load_a16(A + kc * 32, D_, lane);
        v16h w = load_b16(Bt + kc * 32, D_, lane);
        c = wmma16(a, w, c);
    }
    const int col = nt * 16 + (lane & 15);
    const float bv = (dir ? bihB : bihF)[col];
    float* out = dir ? xpB : xpF;
#pragma unroll
    for (int r = 0; r < 8; ++r) {
        int m = m0 + r + ((lane & 16) >> 1);
        out[(size_t)m * G3_ + col] = c[r] + bv;
    }
}

// ---------------- dense conv-as-GEMM, LDS-staged B (TDM), dbl-buffer -------
// 8 waves/block = 8 M-tiles; every wave computes all 4 N-tiles so the B panel
// (64ch x KB_ K-values = 16 KB f16) staged in LDS is reused by 128 output rows.
__global__ __launch_bounds__(256) void conv_gemm_kernel(
    const _Float16* __restrict__ xpad, const _Float16* __restrict__ Wd,
    const float* __restrict__ cbias, _Float16* __restrict__ cnn) {
    __shared__ alignas(16) _Float16 Bsh[2][64 * KB_];
    const int tid  = threadIdx.x;
    const int wave = tid >> 5;
    const int lane = tid & 31;
    const int mt = blockIdx.x * 8 + wave;
    const int m0 = mt * 16;
    const int b  = m0 >> 9;
    const int s0 = m0 & 511;
    // A row for (b,s) = x_pad[b, s .. s+62, :] flattened => contiguous window.
    const _Float16* A = xpad + ((size_t)(b * SP_ + s0)) * D_;

#if USE_TDM_
    if (tid < 32) {                       // wave 0 drives the tensor DMA
        tdm_stage_b(Wd, (unsigned)(uintptr_t)&Bsh[0][0]);
        __builtin_amdgcn_s_wait_tensorcnt(0);
    }
    __syncthreads();
#else
    {
        f32x4 stg[4];
        stage_fb_load(Wd, tid, stg);
        stage_fb_store(&Bsh[0][0], tid, stg);
        __syncthreads();
    }
#endif

    v8f acc[4] = {v8f{}, v8f{}, v8f{}, v8f{}};
    for (int st = 0; st < NSTAGE_; ++st) {
        const int cur = st & 1;
        const bool more = (st + 1) < NSTAGE_;
#if USE_TDM_
        if (more && tid < 32)
            tdm_stage_b(Wd + (size_t)(st + 1) * KB_,
                        (unsigned)(uintptr_t)&Bsh[cur ^ 1][0]);
#else
        f32x4 stg[4];
        if (more) stage_fb_load(Wd + (size_t)(st + 1) * KB_, tid, stg);
#endif
        const _Float16* Bl = &Bsh[cur][0];
        const int k0 = st * KB_;
        __builtin_prefetch(A + k0 + 2 * KB_, 0, 0);    // stream A ahead
#pragma unroll
        for (int kc = 0; kc < KB_ / 32; ++kc) {
            v16h a = load_a16(A + k0 + kc * 32, D_, lane);
#pragma unroll
            for (int nt = 0; nt < 4; ++nt) {
                v16h w = load_b16(Bl + (size_t)nt * 16 * KB_ + kc * 32, KB_, lane);
                acc[nt] = wmma16(a, w, acc[nt]);
            }
        }
        if (more) {
#if USE_TDM_
            if (tid < 32) __builtin_amdgcn_s_wait_tensorcnt(0);
#else
            stage_fb_store(&Bsh[cur ^ 1][0], tid, stg);
#endif
            __syncthreads();
        }
    }

    const int colb = lane & 15;
#pragma unroll
    for (int nt = 0; nt < 4; ++nt) {
        const int col = nt * 16 + colb;
        const float bv = cbias[col];
#pragma unroll
        for (int r = 0; r < 8; ++r) {
            int m = m0 + r + ((lane & 16) >> 1);
            cnn[(size_t)m * 64 + col] = (_Float16)tanhf(acc[nt][r] + bv);
        }
    }
}

// ---------------- GRU sequential scan (batched M=16, WMMA per step) --------
__global__ __launch_bounds__(128) void gru_scan_kernel(
    const float* __restrict__ xp,    // (B,S,192) for this direction
    const float* __restrict__ Whh,   // (192,64) f32
    const float* __restrict__ bhh,   // (192)
    _Float16* __restrict__ hout,     // (B,S,64) f16
    int dir) {
    const int mt   = blockIdx.x;     // batch tile 0..3
    const int wave = threadIdx.x >> 5;
    const int lane = threadIdx.x & 31;
    __shared__ alignas(16) _Float16 hsh[16 * 64];

    for (int i = threadIdx.x; i < 16 * 64; i += 128) hsh[i] = (_Float16)0.f;

    // Preload Whh B-operands for this wave's gate tiles (r,z,n at cols 16w..).
    const int col    = lane & 15;
    const int kbase  = (lane & 16) ? 8 : 0;
    v16h Bop[3][2];
#pragma unroll
    for (int g = 0; g < 3; ++g) {
        const int ng = (wave + 4 * g) * 16 + col;
#pragma unroll
        for (int kc = 0; kc < 2; ++kc) {
            _Float16 tmp[16];
#pragma unroll
            for (int j = 0; j < 16; ++j)
                tmp[j] = (_Float16)Whh[(size_t)ng * H_ + kc * 32 + kbase + (j < 8 ? j : j + 8)];
            Bop[g][kc] = *reinterpret_cast<v16h*>(tmp);
        }
    }
    const float bR = bhh[0 * H_ + wave * 16 + col];
    const float bZ = bhh[1 * H_ + wave * 16 + col];
    const float bN = bhh[2 * H_ + wave * 16 + col];

    v8f h = {};                       // this wave's 16 h-columns, D layout
    const int rowoff = (lane & 16) >> 1;
    const int b0 = mt * 16;
    __syncthreads();

    for (int step = 0; step < S_; ++step) {
        const int s = dir ? (S_ - 1 - step) : step;
        const int arow  = lane & 15;
        const int akoff = (lane & 16) ? 8 : 0;
        v16h a0, a1;
        {
            const _Float16* p = &hsh[arow * 64 + akoff];
            f32x4* v = reinterpret_cast<f32x4*>(&a0);
            v[0] = *reinterpret_cast<const f32x4*>(p);
            v[1] = *reinterpret_cast<const f32x4*>(p + 16);
            p = &hsh[arow * 64 + 32 + akoff];
            v = reinterpret_cast<f32x4*>(&a1);
            v[0] = *reinterpret_cast<const f32x4*>(p);
            v[1] = *reinterpret_cast<const f32x4*>(p + 16);
        }
        __syncthreads();              // all reads done before writes below

        v8f hr = {}, hz = {}, hn = {};
        hr = wmma16(a0, Bop[0][0], hr); hr = wmma16(a1, Bop[0][1], hr);
        hz = wmma16(a0, Bop[1][0], hz); hz = wmma16(a1, Bop[1][1], hz);
        hn = wmma16(a0, Bop[2][0], hn); hn = wmma16(a1, Bop[2][1], hn);

#pragma unroll
        for (int r = 0; r < 8; ++r) {
            const int bb = b0 + r + rowoff;
            const size_t base = ((size_t)bb * S_ + s) * G3_;
            const float xr = xp[base + 0 * H_ + wave * 16 + col];
            const float xz = xp[base + 1 * H_ + wave * 16 + col];
            const float xn = xp[base + 2 * H_ + wave * 16 + col];
            const float rg = 1.f / (1.f + __expf(-(xr + hr[r] + bR)));
            const float zg = 1.f / (1.f + __expf(-(xz + hz[r] + bZ)));
            const float ng = tanhf(xn + rg * (hn[r] + bN));
            const float hv = (1.f - zg) * ng + zg * h[r];
            h[r] = hv;
            const _Float16 hh = (_Float16)hv;
            hsh[(r + rowoff) * 64 + wave * 16 + col] = hh;
            hout[((size_t)bb * S_ + s) * 64 + wave * 16 + col] = hh;
        }
        __syncthreads();              // writes visible before next reads
    }
}

// ---------------- emissions: em = [hF,hB,cnn] @ linW^T + lin_b -------------
__global__ __launch_bounds__(32) void emit_kernel(
    const _Float16* __restrict__ hF, const _Float16* __restrict__ hB,
    const _Float16* __restrict__ cnn, const _Float16* __restrict__ linWh,
    const float* __restrict__ lin_b, float* __restrict__ em) {
    const int mt = blockIdx.x;        // 0..2047
    const int lane = threadIdx.x;
    const int m0 = mt * 16;
    const _Float16* srcs[3] = {hF, hB, cnn};
    v8f c = {};
#pragma unroll
    for (int g = 0; g < 3; ++g)
#pragma unroll
        for (int kc = 0; kc < 2; ++kc) {
            v16h a = load_a16(srcs[g] + (size_t)m0 * 64 + kc * 32, 64, lane);
            v16h w = load_b16(linWh + g * 64 + kc * 32, G3_, lane);
            c = wmma16(a, w, c);
        }
    const int col = lane & 15;
    const float bv = (col < NL_) ? lin_b[col] : 0.f;
#pragma unroll
    for (int r = 0; r < 8; ++r) {
        int m = m0 + r + ((lane & 16) >> 1);
        em[(size_t)m * 16 + col] = c[r] + bv;
    }
}

// ---------------- CRF loss: one wave per batch row -------------------------
__global__ void zero_out_kernel(float* o) {
    if (blockIdx.x == 0 && threadIdx.x == 0) *o = 0.f;
}

__global__ __launch_bounds__(32) void crf_kernel(
    const float* __restrict__ em,     // (B*S, 16)
    const int* __restrict__ label,    // (B, S)
    const float* __restrict__ cstart, const float* __restrict__ cend,
    const float* __restrict__ ctrans, // (9,9)
    float* __restrict__ out) {
    const int b = blockIdx.x;
    const int lane = threadIdx.x;
    const size_t base = (size_t)b * S_;
    const int* lab = label + base;

    // numerator (mask is all-true in this workload; length == S)
    float num = 0.f;
    for (int s = lane; s < S_; s += 32) {
        const int ls = lab[s];
        const float e = em[(base + s) * 16 + ls];
        num += (s == 0) ? (cstart[ls] + e) : (ctrans[lab[s - 1] * NL_ + ls] + e);
    }
#pragma unroll
    for (int off = 16; off > 0; off >>= 1) num += __shfl_down(num, off, 32);

    // denominator: forward algorithm, lane j owns state j (j < 9)
    const int j = lane;
    float tj[NL_];
#pragma unroll
    for (int i = 0; i < NL_; ++i)
        tj[i] = (j < NL_) ? ctrans[i * NL_ + j] : 0.f;
    float alpha = (j < NL_) ? (cstart[j] + em[base * 16 + j]) : -1e30f;
    for (int s = 1; s < S_; ++s) {
        float ai[NL_];
#pragma unroll
        for (int i = 0; i < NL_; ++i) ai[i] = __shfl(alpha, i, 32);
        float m = -1e30f;
#pragma unroll
        for (int i = 0; i < NL_; ++i) m = fmaxf(m, ai[i] + tj[i]);
        float ssum = 0.f;
#pragma unroll
        for (int i = 0; i < NL_; ++i) ssum += __expf(ai[i] + tj[i] - m);
        const float anew = m + __logf(ssum) + em[(base + s) * 16 + j];
        alpha = (j < NL_) ? anew : alpha;
    }
    float v = (j < NL_) ? (alpha + cend[j]) : -1e30f;
    float mx = v;
#pragma unroll
    for (int off = 16; off > 0; off >>= 1) mx = fmaxf(mx, __shfl_xor(mx, off, 32));
    float se = (j < NL_) ? __expf(v - mx) : 0.f;
#pragma unroll
    for (int off = 16; off > 0; off >>= 1) se += __shfl_xor(se, off, 32);
    if (lane == 0) {
        const float denom = mx + __logf(se);
        const float tnum = num + cend[lab[S_ - 1]];
        atomicAdd(out, (denom - tnum) * (1.0f / (float)B_));   // mean(denom-num)
    }
}

// ---------------- host-side orchestration ----------------
static inline size_t alignup(size_t x) { return (x + 255) & ~(size_t)255; }

extern "C" void kernel_launch(void* const* d_in, const int* in_sizes, int n_in,
                              void* d_out, int out_size, void* d_ws, size_t ws_size,
                              hipStream_t stream) {
    (void)in_sizes; (void)out_size; (void)ws_size;

    const float* word2vec = (const float*)d_in[0];
    const int*   label    = (const int*)d_in[3];
    const float* WihF = (const float*)d_in[4];
    const float* WhhF = (const float*)d_in[5];
    const float* bihF = (const float*)d_in[6];
    const float* bhhF = (const float*)d_in[7];
    const float* WihB = (const float*)d_in[8];
    const float* WhhB = (const float*)d_in[9];
    const float* bihB = (const float*)d_in[10];
    const float* bhhB = (const float*)d_in[11];

    Ptr32 cws, cbs;
    const float *linW, *linb, *cstart, *cend, *ctrans;
    if (n_in >= 81) {   // tuple expanded into 32 entries each
        for (int k = 0; k < 32; ++k) {
            cws.p[k] = (const float*)d_in[12 + k];
            cbs.p[k] = (const float*)d_in[44 + k];
        }
        linW = (const float*)d_in[76]; linb = (const float*)d_in[77];
        cstart = (const float*)d_in[78]; cend = (const float*)d_in[79];
        ctrans = (const float*)d_in[80];
    } else {            // tuples flattened to one concatenated buffer each
        const float* wc = (const float*)d_in[12];
        const float* bc = (const float*)d_in[13];
        for (int k = 0; k < 32; ++k) {
            cws.p[k] = wc + (size_t)1536 * k * k;   // prefix of 2*768*(2j+1)
            cbs.p[k] = bc + 2 * k;
        }
        linW = (const float*)d_in[14]; linb = (const float*)d_in[15];
        cstart = (const float*)d_in[16]; cend = (const float*)d_in[17];
        ctrans = (const float*)d_in[18];
    }

    // workspace carve-up
    char* ws = (char*)d_ws;
    size_t off = 0;
    _Float16* xpad  = (_Float16*)(ws + off); off = alignup(off + (size_t)B_ * SP_ * D_ * 2);
    _Float16* Wd    = (_Float16*)(ws + off); off = alignup(off + (size_t)64 * KTOT_ * 2);
    _Float16* WihFh = (_Float16*)(ws + off); off = alignup(off + (size_t)G3_ * D_ * 2);
    _Float16* WihBh = (_Float16*)(ws + off); off = alignup(off + (size_t)G3_ * D_ * 2);
    _Float16* linWh = (_Float16*)(ws + off); off = alignup(off + (size_t)16 * G3_ * 2);
    float*    cbias = (float*)   (ws + off); off = alignup(off + 64 * 4);
    float*    xpF   = (float*)   (ws + off); off = alignup(off + (size_t)B_ * S_ * G3_ * 4);
    float*    xpB   = (float*)   (ws + off); off = alignup(off + (size_t)B_ * S_ * G3_ * 4);
    _Float16* hF    = (_Float16*)(ws + off); off = alignup(off + (size_t)B_ * S_ * H_ * 2);
    _Float16* hB    = (_Float16*)(ws + off); off = alignup(off + (size_t)B_ * S_ * H_ * 2);
    _Float16* cnn   = (_Float16*)(ws + off); off = alignup(off + (size_t)B_ * S_ * H_ * 2);
    float*    em    = (float*)   (ws + off); off = alignup(off + (size_t)B_ * S_ * 16 * 4);

    // 1. pad + f16-convert x
    {
        size_t n = (size_t)B_ * SP_ * D_;
        pad_convert_x_kernel<<<dim3((unsigned)((n + 255) / 256)), dim3(256), 0, stream>>>(
            word2vec, xpad);
    }
    // 2. pack dense conv weights + misc weights
    {
        size_t n = (size_t)64 * KTOT_;
        pack_wd_kernel<<<dim3((unsigned)((n + 255) / 256)), dim3(256), 0, stream>>>(cws, Wd);
        size_t m = (size_t)2 * G3_ * D_ + 16 * G3_ + 64;
        pack_misc_kernel<<<dim3((unsigned)((m + 255) / 256)), dim3(256), 0, stream>>>(
            WihF, WihB, linW, cbs, WihFh, WihBh, linWh, cbias);
    }
    // 3. GRU input projections (both directions), WMMA GEMM
    gru_xproj_kernel<<<dim3(2048, 12, 2), dim3(32), 0, stream>>>(
        xpad, WihFh, WihBh, bihF, bihB, xpF, xpB);
    // 4. dense conv GEMM (TDM-staged LDS B panel, double-buffered) + tanh
    conv_gemm_kernel<<<dim3(256), dim3(256), 0, stream>>>(xpad, Wd, cbias, cnn);
    // 5. GRU recurrences (fwd / bwd)
    gru_scan_kernel<<<dim3(4), dim3(128), 0, stream>>>(xpF, WhhF, bhhF, hF, 0);
    gru_scan_kernel<<<dim3(4), dim3(128), 0, stream>>>(xpB, WhhB, bhhB, hB, 1);
    // 6. emissions GEMM
    emit_kernel<<<dim3(2048), dim3(32), 0, stream>>>(hF, hB, cnn, linWh, linb, em);
    // 7. CRF loss
    zero_out_kernel<<<dim3(1), dim3(32), 0, stream>>>((float*)d_out);
    crf_kernel<<<dim3(B_), dim3(32), 0, stream>>>(em, label, cstart, cend, ctrans,
                                                  (float*)d_out);
}
